// LSTM_Unit_83897891160514
// MI455X (gfx1250) — compile-verified, matrix operand order checked by reading, not run on hardware
//
#include <hip/hip_runtime.h>
#include <hip/hip_fp16.h>
#include <math.h>

typedef __attribute__((ext_vector_type(16))) _Float16 v16h;
typedef __attribute__((ext_vector_type(8)))  float    v8f;
typedef __attribute__((ext_vector_type(4)))  float    v4f;

#define E_DIM 256
#define H_DIM 256
#define B_DIM 65536
#define KBLK  16          // 512 / 32 k-blocks
#define NBLK  64          // 4H / 16 n-blocks
#define MW    4           // 16-row tiles per wave (64 rows / wave)

// ---------------------------------------------------------------------------
// Pack [W_f|W_i|W_o|W_c ; U_f|U_i|U_o|U_c] (fp32, row-major EuH x H each) into
// f16 B-fragments laid out exactly as V_WMMA_F32_16X16X32_F16 consumes them:
//   fragment(kb, nb) -> 32 lanes x 16 halves, contiguous 32B per lane.
//   lanes 0-15 : col N = lane,     K offsets {0..7, 16..23}
//   lanes 16-31: col N = lane-16,  K offsets {8..15, 24..31}
// ---------------------------------------------------------------------------
__global__ __launch_bounds__(256)
void pack_weights_kernel(const float* __restrict__ Wf, const float* __restrict__ Wi,
                         const float* __restrict__ Wo, const float* __restrict__ Wc,
                         const float* __restrict__ Uf, const float* __restrict__ Ui,
                         const float* __restrict__ Uo, const float* __restrict__ Uc,
                         _Float16* __restrict__ wp)
{
    int idx  = blockIdx.x * blockDim.x + threadIdx.x;   // [0, KBLK*NBLK*32*16)
    int i    = idx & 15;          // half index within lane fragment
    int lane = (idx >> 4) & 31;
    int nb   = (idx >> 9) & (NBLK - 1);
    int kb   = idx >> 15;         // 0..15

    int gate = nb >> 4;                                 // 0:f 1:i 2:o 3:c
    int n    = ((nb & 15) << 4) | (lane & 15);          // column within gate's H
    int koff = (lane < 16) ? ((i < 8) ? i      : i + 8)
                           : ((i < 8) ? i + 8  : i + 16);
    int k    = kb * 32 + koff;                          // 0..511 combined K

    const float* Wg = (gate == 0) ? Wf : (gate == 1) ? Wi : (gate == 2) ? Wo : Wc;
    const float* Ug = (gate == 0) ? Uf : (gate == 1) ? Ui : (gate == 2) ? Uo : Uc;
    float v = (k < E_DIM) ? Wg[(size_t)k * H_DIM + n]
                          : Ug[(size_t)(k - E_DIM) * H_DIM + n];
    wp[idx] = (_Float16)v;
}

// ---------------------------------------------------------------------------
// Fused LSTM cell: z = [x|h] @ [W;U] + b via f16 WMMA (f32 accum), gates fused
// in the epilogue. One wave: 64 batch rows x 16 H-cols, all 4 gates.
// ---------------------------------------------------------------------------
__global__ __launch_bounds__(256)
void lstm_wmma_kernel(const float* __restrict__ x, const float* __restrict__ cprev,
                      const float* __restrict__ h,
                      const float* __restrict__ bf, const float* __restrict__ bi,
                      const float* __restrict__ bo, const float* __restrict__ bc,
                      const _Float16* __restrict__ wp,
                      float* __restrict__ cnew, float* __restrict__ hnew)
{
    const int lane    = threadIdx.x & 31;
    const int wave    = threadIdx.x >> 5;
    const int gwave   = blockIdx.x * 8 + wave;
    const int nbh     = gwave & 15;             // which 16-col tile of H
    const int rowg    = gwave >> 4;             // 1024 groups of 64 rows
    const int mbase   = rowg * (16 * MW);
    const int halfsel = lane >> 4;              // 0: lanes 0-15, 1: lanes 16-31
    const int lrow    = lane & 15;

    v8f acc[4][MW];
#pragma unroll
    for (int g = 0; g < 4; ++g)
#pragma unroll
        for (int mt = 0; mt < MW; ++mt)
            acc[g][mt] = (v8f){0.f, 0.f, 0.f, 0.f, 0.f, 0.f, 0.f, 0.f};

    for (int kb = 0; kb < KBLK; ++kb) {
        // --- B fragments: one 32B contiguous load per lane per gate (L2-hot)
        v16h bfrag[4];
#pragma unroll
        for (int g = 0; g < 4; ++g) {
            const v16h* p = (const v16h*)(wp +
                ((((size_t)kb * NBLK + (g * 16 + nbh)) * 32 + lane) << 4));
            bfrag[g] = *p;
        }

        // --- A fragments: fp32 from x (kb<8) or h (kb>=8), cvt to f16
        const float* src = (kb < 8) ? x : h;
        const int kcol = ((kb & 7) * 32) + (halfsel ? 8 : 0);
#pragma unroll
        for (int mt = 0; mt < MW; ++mt) {
            const int row = mbase + mt * 16 + lrow;
            const float* rp = src + (size_t)row * E_DIM + kcol;
            v4f a0 = *(const v4f*)(rp);
            v4f a1 = *(const v4f*)(rp + 4);
            v4f a2 = *(const v4f*)(rp + 16);
            v4f a3 = *(const v4f*)(rp + 20);
            v16h af;
#pragma unroll
            for (int t = 0; t < 4; ++t) {
                af[t]      = (_Float16)a0[t];
                af[4 + t]  = (_Float16)a1[t];
                af[8 + t]  = (_Float16)a2[t];
                af[12 + t] = (_Float16)a3[t];
            }
#pragma unroll
            for (int g = 0; g < 4; ++g)
                acc[g][mt] = __builtin_amdgcn_wmma_f32_16x16x32_f16(
                    /*neg_a=*/false, af, /*neg_b=*/false, bfrag[g],
                    /*c_mod=*/(short)0, acc[g][mt],
                    /*reuse_a=*/false, /*reuse_b=*/false);
        }
    }

    // --- Epilogue: fused gates. C/D layout: VGPR j holds row j + 8*halfsel,
    //     col = lane % 16 within the tile.
    const int col = nbh * 16 + lrow;
    const float Bf = bf[col], Bi = bi[col], Bo = bo[col], Bc = bc[col];
#pragma unroll
    for (int mt = 0; mt < MW; ++mt) {
#pragma unroll
        for (int j = 0; j < 8; ++j) {
            const int row = mbase + mt * 16 + halfsel * 8 + j;
            const float zf = acc[0][mt][j] + Bf;
            const float zi = acc[1][mt][j] + Bi;
            const float zo = acc[2][mt][j] + Bo;
            const float zg = acc[3][mt][j] + Bc;
            const float fg = 1.0f / (1.0f + __expf(-zf));
            const float ig = 1.0f / (1.0f + __expf(-zi));
            const float og = 1.0f / (1.0f + __expf(-zo));
            const float gg = tanhf(zg);
            const size_t off = (size_t)row * H_DIM + col;
            const float cn = cprev[off] * fg + gg * ig;
            cnew[off] = cn;
            hnew[off] = og * tanhf(cn);
        }
    }
}

extern "C" void kernel_launch(void* const* d_in, const int* in_sizes, int n_in,
                              void* d_out, int out_size, void* d_ws, size_t ws_size,
                              hipStream_t stream) {
    // setup_inputs order: x, c, h, W_i, U_i, b_i, W_f, U_f, b_f, W_c, U_c, b_c, W_o, U_o, b_o
    const float* x   = (const float*)d_in[0];
    const float* c   = (const float*)d_in[1];
    const float* h   = (const float*)d_in[2];
    const float* W_i = (const float*)d_in[3];
    const float* U_i = (const float*)d_in[4];
    const float* b_i = (const float*)d_in[5];
    const float* W_f = (const float*)d_in[6];
    const float* U_f = (const float*)d_in[7];
    const float* b_f = (const float*)d_in[8];
    const float* W_c = (const float*)d_in[9];
    const float* U_c = (const float*)d_in[10];
    const float* b_c = (const float*)d_in[11];
    const float* W_o = (const float*)d_in[12];
    const float* U_o = (const float*)d_in[13];
    const float* b_o = (const float*)d_in[14];

    _Float16* wp = (_Float16*)d_ws;            // 1 MB packed f16 weights
    float* cnew  = (float*)d_out;              // tuple order: (c_new, h_new)
    float* hnew  = cnew + (size_t)B_DIM * H_DIM;

    // 1) pack weights into WMMA B-fragment layout (gate order f,i,o,c)
    const int pack_elems = KBLK * NBLK * 32 * 16;   // 524288 halves
    pack_weights_kernel<<<pack_elems / 256, 256, 0, stream>>>(
        W_f, W_i, W_o, W_c, U_f, U_i, U_o, U_c, wp);

    // 2) fused WMMA GEMM + gates: 16384 waves = 2048 blocks x 8 waves
    const int total_waves = (B_DIM / (16 * MW)) * (H_DIM / 16);
    lstm_wmma_kernel<<<total_waves / 8, 256, 0, stream>>>(
        x, c, h, b_f, b_i, b_o, b_c, wp, cnew, hnew);
}
